// PlotLine_16260746183106
// MI455X (gfx1250) — matrix operand order repaired and checked by reading, not compiled, not run on hardware
//
#include <hip/hip_runtime.h>
#include <hip/hip_bf16.h>
#include <math.h>

// ---------------- problem constants (from reference) ----------------
#define IMG        128                 // IMG_SIZE
#define NSEG       31                  // P-1
#define CMP        181                 // int(128*sqrt(2))
#define LTOT       (NSEG * CMP)        // 5611 line samples per image
#define NIMG       8
#define KCHUNK     32                  // WMMA K per step (f16 16x16x32)
#define NCHUNK     ((LTOT + KCHUNK - 1) / KCHUNK)   // 176
#define KSPLIT     22                  // K-split blocks per image
#define CPB        (NCHUNK / KSPLIT)   // 8 chunks per block (22*8 = 176 exactly)
#define NOUT       (NIMG * IMG * IMG)  // 131072 f32 outputs

typedef __attribute__((ext_vector_type(16))) _Float16 v16h;
typedef __attribute__((ext_vector_type(8)))  float    v8f;

static __device__ __forceinline__ unsigned int pack_f16x2(float a, float b) {
    _Float16 ha = (_Float16)a, hb = (_Float16)b;
    unsigned short ua = __builtin_bit_cast(unsigned short, ha);
    unsigned short ub = __builtin_bit_cast(unsigned short, hb);
    return (unsigned int)ua | ((unsigned int)ub << 16);
}

// ---------------- prologue: zero the accumulation target ----------------
__global__ void zero_out_kernel(float* __restrict__ out) {
    int i = blockIdx.x * blockDim.x + threadIdx.x;
    if (i < NOUT) out[i] = 0.0f;
}

// ---------------- main WMMA GEMM kernel ----------------
// grid = (NIMG, KSPLIT), block = 256 (8 wave32).
// Block (n, s) accumulates chunks [s*CPB, (s+1)*CPB) of the K dimension into
// out[n] via f32 atomics. D[i,j] = sum_l exp(-(x_l-i)^2/2w) * exp(-(y_l-j)^2/2w).
__global__ void __launch_bounds__(256)
plotline_wmma_kernel(const float* __restrict__ pts, float* __restrict__ out) {
    const int tid  = threadIdx.x;
    const int lane = tid & 31;        // wave32 lane
    const int wave = tid >> 5;        // 8 waves = 8 row-tiles of 16
    const int n    = blockIdx.x;      // image
    const int c0   = blockIdx.y * CPB;

    __shared__ float s_pts[32 * 3];   // control points of this image
    __shared__ float s_x[KCHUNK], s_y[KCHUNK], s_iw[KCHUNK];
    // Operands staged in the exact per-lane WMMA register layout:
    // s_A[rowTile][lane][dword], s_B[colTile][lane][dword]; dword = packed f16x2.
    __shared__ __align__(32) unsigned int s_A[8][32][8];
    __shared__ __align__(32) unsigned int s_B[8][32][8];

    if (tid < 96) s_pts[tid] = pts[n * 96 + tid];

    v8f acc[8] = {};                  // 8 col-tiles x (16x16 f32) per wave

    for (int c = c0; c < c0 + CPB; ++c) {
        __syncthreads();              // prior reads of s_* finished (also covers s_pts fill)

        // ---- stage 1: 32 sample params for this K-chunk ----
        if (tid < KCHUNK) {
            int l = c * KCHUNK + tid;
            float x, y, iw;
            if (l < LTOT) {
                int   p   = l / CMP;            // segment 0..30
                int   ti  = l - p * CMP;        // 0..180
                float tt  = (float)ti * (1.0f / (float)CMP);
                float omt = 1.0f - tt;
                const float* P0 = &s_pts[p * 3];
                x = omt * P0[0] + tt * P0[3];
                y = omt * P0[1] + tt * P0[4];
                float w = omt * P0[2] + tt * P0[5];
                iw = 0.5f / w;                  // 1/(2w)
            } else {                            // tail: force exp(...) == 0
                x = 1.0e9f; y = 1.0e9f; iw = 1.0f;
            }
            s_x[tid] = x; s_y[tid] = y; s_iw[tid] = iw;
        }
        __syncthreads();

        // ---- stage 2: cooperatively build A (128x32) and B (32x128) as f16 ----
        // Thread writes dword v = tid&7 of lane tid>>3 in every tile -> each exp
        // computed exactly once per block.
        {
            const int l8  = tid >> 3;           // 0..31 (target lane)
            const int v   = tid & 7;            // 0..7  (target dword)
            const int hi  = l8 >> 4;            // lane-group
            const int mn  = l8 & 15;
            // A layout (ISA 16-bit A 16x32): k = (v&3)*2+h + hi*8 + (v>>2)*16
            const int ka  = ((v & 3) << 1) + (hi << 3) + ((v >> 2) << 4);
            // B layout (matches A): n = lane&15, k = (v&3)*2+h + hi*8 + (v>>2)*16
            const int kb  = ((v & 3) << 1) + (hi << 3) + ((v >> 2) << 4);
#pragma unroll
            for (int rt = 0; rt < 8; ++rt) {    // 8 row tiles of A
                float m  = (float)(rt * 16 + mn);
                float d0 = s_x[ka]     - m;
                float d1 = s_x[ka + 1] - m;
                float e0 = __expf(-d0 * d0 * s_iw[ka]);
                float e1 = __expf(-d1 * d1 * s_iw[ka + 1]);
                s_A[rt][l8][v] = pack_f16x2(e0, e1);
            }
#pragma unroll
            for (int ct = 0; ct < 8; ++ct) {    // 8 col tiles of B
                float nn = (float)(ct * 16 + mn);
                float d0 = s_y[kb]     - nn;
                float d1 = s_y[kb + 1] - nn;
                float e0 = __expf(-d0 * d0 * s_iw[kb]);
                float e1 = __expf(-d1 * d1 * s_iw[kb + 1]);
                s_B[ct][l8][v] = pack_f16x2(e0, e1);
            }
        }
        __syncthreads();

        // ---- stage 3: 8 WMMAs per wave (row strip x all 8 col tiles) ----
        const v16h a = *reinterpret_cast<const v16h*>(&s_A[wave][lane][0]);
#pragma unroll
        for (int ct = 0; ct < 8; ++ct) {
            const v16h b = *reinterpret_cast<const v16h*>(&s_B[ct][lane][0]);
            acc[ct] = __builtin_amdgcn_wmma_f32_16x16x32_f16(
                false, a, false, b, (short)0, acc[ct], false, false);
        }
    }

    // ---- epilogue: f32 atomic accumulation into out[n] ----
    // C/D layout: VGPR g -> row = waveTile*16 + g + 8*(lane>=16), col = tile*16 + (lane&15)
    float* outn = out + n * IMG * IMG;
    const int rowbase = wave * 16 + ((lane >> 4) << 3);
    const int colbase = lane & 15;
#pragma unroll
    for (int ct = 0; ct < 8; ++ct) {
#pragma unroll
        for (int g = 0; g < 8; ++g) {
            atomicAdd(&outn[(rowbase + g) * IMG + ct * 16 + colbase], acc[ct][g]);
        }
    }
}

// ---------------- epilogue: tanh in place ----------------
__global__ void tanh_kernel(float* __restrict__ out) {
    int i = blockIdx.x * blockDim.x + threadIdx.x;
    if (i < NOUT) out[i] = tanhf(out[i]);
}

extern "C" void kernel_launch(void* const* d_in, const int* in_sizes, int n_in,
                              void* d_out, int out_size, void* d_ws, size_t ws_size,
                              hipStream_t stream) {
    (void)in_sizes; (void)n_in; (void)out_size; (void)d_ws; (void)ws_size;
    const float* pts = (const float*)d_in[0];     // (8,32,3) f32
    float* out = (float*)d_out;                   // (8,128,128) f32

    zero_out_kernel<<<(NOUT + 255) / 256, 256, 0, stream>>>(out);
    plotline_wmma_kernel<<<dim3(NIMG, KSPLIT), 256, 0, stream>>>(pts, out);
    tanh_kernel<<<(NOUT + 255) / 256, 256, 0, stream>>>(out);
}
